// VectorQuantizer_45277545234651
// MI455X (gfx1250) — matrix operand order repaired and checked by reading, not compiled, not run on hardware
//
#include <hip/hip_runtime.h>

// VectorQuantizer forward for MI455X (gfx1250, wave32, WMMA).
// B=32, D=64, T=8192, K=512.  dist = ||x||^2 - 2 x.e + ||e||^2 ; argmin over k
// is unaffected by the ||x||^2 term, so we compute (e2[k] - 2*xe) with xe done
// in f16 WMMA (f32 accumulate) and e2 in exact f32.

typedef _Float16 v16h __attribute__((ext_vector_type(16)));
typedef _Float16 h8   __attribute__((ext_vector_type(8)));
typedef _Float16 h4   __attribute__((ext_vector_type(4)));
typedef float    v8f  __attribute__((ext_vector_type(8)));
typedef float    f4   __attribute__((ext_vector_type(4)));

#define B_ 32
#define D_ 64
#define T_ 8192
#define K_ 512

__global__ __launch_bounds__(256) void vq_wmma_45277545234651(
    const float* __restrict__ x,     // [B, D, T]
    const float* __restrict__ emb,   // [K, D]
    float* __restrict__ out)         // [2, B, D, T] (st + quantized, identical fwd)
{
  extern __shared__ char smem[];
  _Float16* sEmb = (_Float16*)smem;             // K_*D_ f16 = 64 KB, layout [k][d]
  float*    sE2  = (float*)(smem + K_ * D_ * 2); // K_ f32 norms = 2 KB

  const int tid = threadIdx.x;

  // ---- cooperative preload: codebook f32->f16 into LDS + exact f32 norms ----
  for (int k = tid; k < K_; k += 256) {
    const f4* row = (const f4*)(emb + k * D_);
    float acc = 0.f;
#pragma unroll
    for (int j = 0; j < D_ / 4; ++j) {
      f4 v = row[j];
      acc = fmaf(v.x, v.x, acc);
      acc = fmaf(v.y, v.y, acc);
      acc = fmaf(v.z, v.z, acc);
      acc = fmaf(v.w, v.w, acc);
      h4 hv = { (_Float16)v.x, (_Float16)v.y, (_Float16)v.z, (_Float16)v.w };
      *(h4*)(sEmb + k * D_ + 4 * j) = hv;
    }
    sE2[k] = acc;
  }
  __syncthreads();

  const int wave = tid >> 5;
  const int lane = tid & 31;
  const int h    = lane >> 4;   // half-wave (0/1)
  const int n    = lane & 15;   // column (t) / A-row (k) within half

  const int tile = blockIdx.x * 8 + wave;   // 16384 t-tiles total
  const int b    = tile >> 9;               // 512 tiles per batch element
  const int t0   = (tile & 511) << 4;

  // ---- load x tile into two B fragments (K=d: 0..31 and 32..63) ----
  // B layout: lane holds column t = t0+n; element j holds K = 8h + j (j<8), 16 + 8h + (j-8) (j>=8).
  const float* px = x + (size_t)b * D_ * T_ + (size_t)(8 * h) * T_ + (t0 + n);
  v16h bf0, bf1;
#pragma unroll
  for (int j = 0; j < 16; ++j) {
    bf0[j] = (_Float16)px[(size_t)(j + (j & 8)) * T_];          // d = 8h + j (+8 for j>=8)
    bf1[j] = (_Float16)px[(size_t)(j + (j & 8) + 32) * T_];   // d = 32 + 8h + j (+8 for j>=8)
  }

  float minv[8];
  int   mini[8];
#pragma unroll
  for (int r = 0; r < 8; ++r) { minv[r] = 3.0e38f; mini[r] = 0; }

  // ---- k loop: 32 tiles of 16 codebook rows; 2 WMMAs each (D=64 = 2*32) ----
  for (int kt = 0; kt < 32; ++kt) {
    // A fragment: lane holds emb row k = kt*16 + n; ISA d-split {8h..8h+7} U {16+8h..}
    const _Float16* arow = sEmb + (kt * 16 + n) * D_ + 8 * h;
    h8 lo0 = *(const h8*)(arow);        // d = 8h + 0..7
    h8 hi0 = *(const h8*)(arow + 16);   // d = 16 + 8h + 0..7
    h8 lo1 = *(const h8*)(arow + 32);   // d = 32 + 8h + 0..7
    h8 hi1 = *(const h8*)(arow + 48);   // d = 48 + 8h + 0..7
    v16h a0 = __builtin_shufflevector(lo0, hi0, 0,1,2,3,4,5,6,7,8,9,10,11,12,13,14,15);
    v16h a1 = __builtin_shufflevector(lo1, hi1, 0,1,2,3,4,5,6,7,8,9,10,11,12,13,14,15);

    v8f c = {0.f, 0.f, 0.f, 0.f, 0.f, 0.f, 0.f, 0.f};
    c = __builtin_amdgcn_wmma_f32_16x16x32_f16(false, a0, false, bf0,
                                               (short)0, c, false, false);
    c = __builtin_amdgcn_wmma_f32_16x16x32_f16(false, a1, false, bf1,
                                               (short)0, c, false, false);

    // C layout: lane holds t = t0+n, VGPR r holds k = kt*16 + 8h + r.
    const float* pe = sE2 + kt * 16 + 8 * h;
    f4 e2a = *(const f4*)pe;
    f4 e2b = *(const f4*)(pe + 4);
#pragma unroll
    for (int r = 0; r < 8; ++r) {
      float e2v  = (r < 4) ? e2a[r] : e2b[r - 4];
      float dist = fmaf(c[r], -2.0f, e2v);   // e2[k] - 2*xe
      int   kg   = kt * 16 + 8 * h + r;
      bool  p    = dist < minv[r];           // strict: first (lowest kt) wins ties
      mini[r] = p ? kg : mini[r];
      minv[r] = p ? dist : minv[r];
    }
  }

  // ---- per-lane reduction over the 8 row-slots (keep lowest index on ties) ----
  float bv = minv[0];
  int   bi = mini[0];
#pragma unroll
  for (int r = 1; r < 8; ++r) {
    bool p = (minv[r] < bv) || ((minv[r] == bv) && (mini[r] < bi));
    bi = p ? mini[r] : bi;
    bv = p ? minv[r] : bv;
  }
  // merge the two half-waves (same t, complementary k mod 16 residues)
  float ov = __shfl_xor(bv, 16, 32);
  int   oi = __shfl_xor(bi, 16, 32);
  {
    bool p = (ov < bv) || ((ov == bv) && (oi < bi));
    bi = p ? oi : bi;
  }

  // ---- gather winning f32 codebook row (L2-hot) and write both outputs ----
  // half 0 writes d = 0..31, half 1 writes d = 32..63 for its t = t0 + n.
  const float* qrow = emb + (size_t)bi * D_ + 32 * h;
  float* o1 = out + (size_t)b * D_ * T_ + (size_t)(32 * h) * T_ + (t0 + n);
  float* o2 = o1 + (size_t)B_ * D_ * T_;
#pragma unroll
  for (int c8 = 0; c8 < 8; ++c8) {
    f4 q = *(const f4*)(qrow + 4 * c8);
#pragma unroll
    for (int j = 0; j < 4; ++j) {
      o1[(size_t)(4 * c8 + j) * T_] = q[j];   // lanes of a half: consecutive t -> 64B
      o2[(size_t)(4 * c8 + j) * T_] = q[j];
    }
  }
}

extern "C" void kernel_launch(void* const* d_in, const int* in_sizes, int n_in,
                              void* d_out, int out_size, void* d_ws, size_t ws_size,
                              hipStream_t stream) {
  (void)in_sizes; (void)n_in; (void)d_ws; (void)ws_size; (void)out_size;
  const float* x   = (const float*)d_in[0];
  const float* emb = (const float*)d_in[1];
  float* out = (float*)d_out;

  const int tiles  = B_ * (T_ / 16);      // 16384
  const int blocks = tiles / 8;           // 8 waves per block, 1 tile per wave
  const size_t shmem = (size_t)K_ * D_ * 2 + (size_t)K_ * 4;  // 66 KB

  vq_wmma_45277545234651<<<blocks, 256, shmem, stream>>>(x, emb, out);
}
